// IGNN_Layer_15693810499839
// MI455X (gfx1250) — matrix operand order repaired and checked by reading, not compiled, not run on hardware
//
#include <hip/hip_runtime.h>
#include <hip/hip_bf16.h>
#include <math.h>

typedef __attribute__((ext_vector_type(16))) __bf16 v16bf;
typedef __attribute__((ext_vector_type(8)))  float  v8f;
typedef __attribute__((ext_vector_type(4)))  __bf16 bf16x4;

#define N_NODES 50000
#define N_EDGES 800000
#define D_H 128
#define D_E 16
#define D_M 128

#define KE   288   // edge GEMM1 K (2*128 h + 1 radial + 16 attr, padded to 9*32)
#define AS   296   // edge A-tile LDS row stride (bf16 elems) - bank-conflict pad
#define MS   136   // m-tile LDS row stride (bf16)
#define M2S  132   // m2 float-tile row stride (f32)
#define KN   256   // node GEMM1 K ([h | msum])
#define ANS  264   // node A-tile row stride (bf16)

#define ME   32    // edges per A-tile (two 16-row WMMA sub-tiles)
#define EB   64    // edges per block (2 tiles, 2 waves, N-split cooperation)

struct alignas(32) Pair { int4 a, b; };

// A-fragment (16x32 bf16, wave32): lane l, row m=l&15, holds K in
// [kb,kb+8) and [kb+16,kb+24) with kb = 8*(l>=16). p points at row+kb.
__device__ __forceinline__ v16bf frag_a(const __bf16* p) {
  Pair q;
  q.a = *(const int4*)(p);
  q.b = *(const int4*)(p + 16);
  return __builtin_bit_cast(v16bf, q);
}
// B-fragment (32x16 bf16): lane l, col n=l&15, holds K in
// [kb,kb+8) and [kb+16,kb+24) with kb = 8*(l>=16). B[k][n] = W[n][k] (row-major W).
__device__ __forceinline__ v16bf frag_b(const __bf16* p) {
  Pair q;
  q.a = *(const int4*)(p);
  q.b = *(const int4*)(p + 16);
  return __builtin_bit_cast(v16bf, q);
}

__device__ __forceinline__ float silu_f(float x) { return x / (1.f + __expf(-x)); }

__device__ __forceinline__ void wave_sync() {
  __builtin_amdgcn_wave_barrier();      // scheduling fence (same-wave LDS is in-order)
  asm volatile("" ::: "memory");
}

struct alignas(16) EdgeLds {
  __bf16 A[ME * AS];     // 18944 B, reused as float m2[ME*M2S] (16896 B) after GEMM2
  __bf16 mT[ME * MS];    // 8704 B
  int    idx[2 * ME];    // rows [0..31], cols [32..63]
  float  att[ME];
};                        // 28032 B ; x2 tiles/block = 56064 B

struct alignas(16) NodeLds {
  __bf16 A[16 * ANS];    // 8448 B
  __bf16 mT[16 * MS];    // 4352 B
};

// ---------------------------------------------------------------- prep / zero
__global__ void prep_weights(const float* __restrict__ We1, const float* __restrict__ We2,
                             const float* __restrict__ Wh1, const float* __restrict__ Wh2,
                             __bf16* __restrict__ ws) {
  int i = blockIdx.x * 256 + threadIdx.x;
  __bf16* W1b  = ws;
  __bf16* W2b  = W1b + 128 * KE;
  __bf16* Wh1b = W2b + 128 * 128;
  __bf16* Wh2b = Wh1b + 128 * KN;
  if (i < 128 * KE) {
    int n = i / KE, k = i % KE;
    W1b[i] = (__bf16)((k < 273) ? We1[n * 273 + k] : 0.f);
  } else if (i < 128 * KE + 128 * 128) {
    int j = i - 128 * KE;
    W2b[j] = (__bf16)We2[j];
  } else if (i < 128 * KE + 128 * 128 + 128 * KN) {
    int j = i - (128 * KE + 128 * 128);
    Wh1b[j] = (__bf16)Wh1[j];
  } else if (i < 128 * KE + 2 * 128 * 128 + 128 * KN) {
    int j = i - (128 * KE + 128 * 128 + 128 * KN);
    Wh2b[j] = (__bf16)Wh2[j];
  }
}

__global__ void zero_out(float* __restrict__ p, int n4) {
  float4 z = {0.f, 0.f, 0.f, 0.f};
  for (int k = blockIdx.x * blockDim.x + threadIdx.x; k < n4; k += gridDim.x * blockDim.x)
    ((float4*)p)[k] = z;
}

// ---------------------------------------------------------------- edge kernel
// Block = 2 waves = 64 edges (two 32-edge A-tiles). Wave w computes output
// N-tiles [4w, 4w+4) for BOTH A-tiles -> each B fragment feeds 4 WMMAs.
__global__ __launch_bounds__(EB) void egnn_edge_kernel(
    const float* __restrict__ x, const float* __restrict__ h,
    const float* __restrict__ edge_attr,
    const float* __restrict__ be1, const float* __restrict__ be2,
    const float* __restrict__ Wa, const float* __restrict__ ba,
    const long long* __restrict__ eidx,
    const __bf16* __restrict__ W1b, const __bf16* __restrict__ W2b,
    float* __restrict__ msum) {
  __shared__ EdgeLds sm[2];
  const int w = threadIdx.x >> 5;     // wave id within block (0/1)
  EdgeLds& MY = sm[w];                // tile this wave builds & scatters
  const int lane = threadIdx.x & 31;
  const int lm = lane & 15, lh = lane >> 4;
  const int nt0 = 4 * w;              // this wave's N-tile range

  // loop-invariant per-lane epilogue constants (only our 4 N-tiles)
  float bb1[4], bb2[4];
#pragma unroll
  for (int ntl = 0; ntl < 4; ++ntl) {
    bb1[ntl] = be1[(nt0 + ntl) * 16 + lm];
    bb2[ntl] = be2[(nt0 + ntl) * 16 + lm];
  }
  const float ba0 = ba[0];

  const int t = blockIdx.x;           // one 64-edge group per block (12500 blocks)
  const int e0 = t * EB + ME * w;     // this wave's 32 edges

  // ---- build this wave's A-tile --------------------------------------------
  MY.idx[lane]      = (int)eidx[(size_t)e0 + lane];
  MY.idx[ME + lane] = (int)eidx[(size_t)N_EDGES + e0 + lane];
  wave_sync();
  {
    const int e = lane;
    const int r = MY.idx[e], c = MY.idx[ME + e];
    float dx = x[3 * r + 0] - x[3 * c + 0];
    float dy = x[3 * r + 1] - x[3 * c + 1];
    float dz = x[3 * r + 2] - x[3 * c + 2];
    float rad = sqrtf(dx * dx + dy * dy + dz * dz);
    __bf16* Ae = &MY.A[e * AS];
    Ae[256] = (__bf16)rad;
    const float* ea = edge_attr + (size_t)(e0 + e) * D_E;
#pragma unroll
    for (int k = 0; k < D_E; ++k) Ae[257 + k] = (__bf16)ea[k];
#pragma unroll
    for (int k = 273; k < KE; ++k) Ae[k] = (__bf16)0.f;
  }
#pragma unroll 4
  for (int i = 0; i < 64; ++i) {      // gather h[row] | h[col] (coalesced float4)
    int f = lane + 32 * i;
    int e = f >> 6, q = f & 63;
    int src = (q < 32) ? MY.idx[e] : MY.idx[ME + e];
    int qq = (q < 32) ? q : (q - 32);
    float4 v = *(const float4*)(h + (size_t)src * D_H + 4 * qq);
    bf16x4 o;
    o.x = (__bf16)v.x; o.y = (__bf16)v.y; o.z = (__bf16)v.z; o.w = (__bf16)v.w;
    *(bf16x4*)&MY.A[e * AS + ((q < 32) ? 0 : 128) + 4 * qq] = o;
  }
  __syncthreads();                    // both A-tiles visible to both waves

  // ---- GEMM1: [64 x 288] x [288 x 64-col-slice]  (9 kt x 4 ntl x 4 mt) -----
  // acc[mt][ntl]: mt 0/1 = sm[0] rows 0-15/16-31, mt 2/3 = sm[1] rows
  v8f acc[4][4] = {};
  for (int kt = 0; kt < KE / 32; ++kt) {
    v16bf af[4];
    af[0] = frag_a(&sm[0].A[lm * AS + kt * 32 + lh * 8]);
    af[1] = frag_a(&sm[0].A[(16 + lm) * AS + kt * 32 + lh * 8]);
    af[2] = frag_a(&sm[1].A[lm * AS + kt * 32 + lh * 8]);
    af[3] = frag_a(&sm[1].A[(16 + lm) * AS + kt * 32 + lh * 8]);
    const __bf16* bp = W1b + (size_t)(nt0 * 16 + lm) * KE + kt * 32 + lh * 8;
    v16bf bf = frag_b(bp);
#pragma unroll
    for (int ntl = 0; ntl < 4; ++ntl) {
      v16bf bnext = bf;               // rotate: prefetch next B before consuming
      if (ntl < 3) bnext = frag_b(bp + (size_t)(ntl + 1) * 16 * KE);
#pragma unroll
      for (int mt = 0; mt < 4; ++mt)
        acc[mt][ntl] = __builtin_amdgcn_wmma_f32_16x16x32_bf16(
            false, af[mt], false, bf, (short)0, acc[mt][ntl], false, false);
      bf = bnext;
    }
  }
  // bias + SiLU -> m-tiles (bf16); each wave writes its column slice of BOTH tiles
#pragma unroll
  for (int ntl = 0; ntl < 4; ++ntl) {
    const int cc = (nt0 + ntl) * 16 + lm;
#pragma unroll
    for (int mt = 0; mt < 4; ++mt) {
      __bf16* mTt = sm[mt >> 1].mT;
      const int rb = (mt & 1) * 16;
#pragma unroll
      for (int j = 0; j < 8; ++j)
        mTt[(rb + j + 8 * lh) * MS + cc] = (__bf16)silu_f(acc[mt][ntl][j] + bb1[ntl]);
    }
  }
  __syncthreads();

  // ---- GEMM2: [64 x 128] x [128 x 64-col-slice]  (4 kt x 4 ntl x 4 mt) -----
  v8f c2[4][4] = {};
  for (int kt = 0; kt < 4; ++kt) {
    v16bf af[4];
    af[0] = frag_a(&sm[0].mT[lm * MS + kt * 32 + lh * 8]);
    af[1] = frag_a(&sm[0].mT[(16 + lm) * MS + kt * 32 + lh * 8]);
    af[2] = frag_a(&sm[1].mT[lm * MS + kt * 32 + lh * 8]);
    af[3] = frag_a(&sm[1].mT[(16 + lm) * MS + kt * 32 + lh * 8]);
    const __bf16* bp = W2b + (size_t)(nt0 * 16 + lm) * D_M + kt * 32 + lh * 8;
    v16bf bf = frag_b(bp);
#pragma unroll
    for (int ntl = 0; ntl < 4; ++ntl) {
      v16bf bnext = bf;
      if (ntl < 3) bnext = frag_b(bp + (size_t)(ntl + 1) * 16 * D_M);
#pragma unroll
      for (int mt = 0; mt < 4; ++mt)
        c2[mt][ntl] = __builtin_amdgcn_wmma_f32_16x16x32_bf16(
            false, af[mt], false, bf, (short)0, c2[mt][ntl], false, false);
      bf = bnext;
    }
  }
  // bias + SiLU -> f32 m2 tiles (reuse A-tile LDS space)
#pragma unroll
  for (int ntl = 0; ntl < 4; ++ntl) {
    const int cc = (nt0 + ntl) * 16 + lm;
#pragma unroll
    for (int mt = 0; mt < 4; ++mt) {
      float* m2t = (float*)(void*)sm[mt >> 1].A;
      const int rb = (mt & 1) * 16;
#pragma unroll
      for (int j = 0; j < 8; ++j)
        m2t[(rb + j + 8 * lh) * M2S + cc] = silu_f(c2[mt][ntl][j] + bb2[ntl]);
    }
  }
  __syncthreads();

  // ---- attention gate + gated scatter (each wave: its own 32 edges) --------
  float* m2 = (float*)(void*)MY.A;
  {
    float dot = ba0;
    const float* mr = m2 + lane * M2S;
#pragma unroll 8
    for (int k = 0; k < D_M; k += 4) {
      float4 mv = *(const float4*)(mr + k);
      float4 wv = *(const float4*)(Wa + k);
      dot += mv.x * wv.x + mv.y * wv.y + mv.z * wv.z + mv.w * wv.w;
    }
    MY.att[lane] = 1.f / (1.f + __expf(-dot));
  }
  wave_sync();
  for (int i = 0; i < 4 * ME; ++i) {
    int e = i >> 2;
    int cc = (i & 3) * 32 + lane;
    float v = m2[e * M2S + cc] * MY.att[e];
    atomicAdd(&msum[(size_t)MY.idx[e] * D_M + cc], v);
  }
}

// ---------------------------------------------------------------- node kernel
__global__ __launch_bounds__(128) void egnn_node_kernel(
    const float* __restrict__ h,
    const float* __restrict__ bh1, const float* __restrict__ bh2,
    const __bf16* __restrict__ Wh1b, const __bf16* __restrict__ Wh2b,
    float* __restrict__ out /* in: message_sum rows, out: final rows */) {
  __shared__ NodeLds sm[4];
  NodeLds& L = sm[threadIdx.x >> 5];
  const int lane = threadIdx.x & 31;
  const int lm = lane & 15, lh = lane >> 4;
  const int wave = blockIdx.x * 4 + (threadIdx.x >> 5);
  const int nw = gridDim.x * 4;

  float bb1[8], bb2[8];
#pragma unroll
  for (int nt = 0; nt < 8; ++nt) {
    bb1[nt] = bh1[nt * 16 + lm];
    bb2[nt] = bh2[nt * 16 + lm];
  }

  for (int t = wave; t < N_NODES / 16; t += nw) {
    const int v0 = t * 16;
    // A = [h | msum] bf16 (contiguous rows, coalesced)
#pragma unroll 4
    for (int i = 0; i < 32; ++i) {
      int f = lane + 32 * i;
      int e = f >> 6, q = f & 63;
      const float* src = (q < 32) ? (h + (size_t)(v0 + e) * D_H + 4 * q)
                                  : (out + (size_t)(v0 + e) * D_H + 4 * (q - 32));
      float4 v = *(const float4*)src;
      bf16x4 o;
      o.x = (__bf16)v.x; o.y = (__bf16)v.y; o.z = (__bf16)v.z; o.w = (__bf16)v.w;
      *(bf16x4*)&L.A[e * ANS + ((q < 32) ? 4 * q : (128 + 4 * (q - 32)))] = o;
    }
    wave_sync();

    // GEMM1: [16 x 256] x [256 x 128]  (8 x 8 = 64 WMMA)
    v8f acc[8] = {};
    for (int kt = 0; kt < KN / 32; ++kt) {
      v16bf af = frag_a(&L.A[lm * ANS + kt * 32 + lh * 8]);
      const __bf16* bbase = Wh1b + (size_t)lm * KN + kt * 32 + lh * 8;
#pragma unroll
      for (int nt = 0; nt < 8; ++nt) {
        v16bf bf = frag_b(bbase + (size_t)nt * 16 * KN);
        acc[nt] = __builtin_amdgcn_wmma_f32_16x16x32_bf16(false, af, false, bf,
                                                          (short)0, acc[nt], false, false);
      }
    }
#pragma unroll
    for (int nt = 0; nt < 8; ++nt) {
      const int cc = nt * 16 + lm;
#pragma unroll
      for (int j = 0; j < 8; ++j)
        L.mT[(j + 8 * lh) * MS + cc] = (__bf16)silu_f(acc[nt][j] + bb1[nt]);
    }
    wave_sync();

    // GEMM2: [16 x 128] x [128 x 128]  (4 x 8 = 32 WMMA)
    v8f acc2[8] = {};
    for (int kt = 0; kt < 4; ++kt) {
      v16bf af = frag_a(&L.mT[lm * MS + kt * 32 + lh * 8]);
      const __bf16* bbase = Wh2b + (size_t)lm * D_M + kt * 32 + lh * 8;
#pragma unroll
      for (int nt = 0; nt < 8; ++nt) {
        v16bf bf = frag_b(bbase + (size_t)nt * 16 * D_M);
        acc2[nt] = __builtin_amdgcn_wmma_f32_16x16x32_bf16(false, af, false, bf,
                                                           (short)0, acc2[nt], false, false);
      }
    }
    // residual + bias, overwrite out rows (msum already consumed)
#pragma unroll
    for (int nt = 0; nt < 8; ++nt) {
      const int cc = nt * 16 + lm;
#pragma unroll
      for (int j = 0; j < 8; ++j) {
        int r = j + 8 * lh;
        size_t off = (size_t)(v0 + r) * D_H + cc;
        out[off] = acc2[nt][j] + bb2[nt] + h[off];
      }
    }
    wave_sync();
  }
}

// ---------------------------------------------------------------- launch
extern "C" void kernel_launch(void* const* d_in, const int* in_sizes, int n_in,
                              void* d_out, int out_size, void* d_ws, size_t ws_size,
                              hipStream_t stream) {
  (void)in_sizes; (void)n_in; (void)out_size; (void)ws_size;
  const float* x   = (const float*)d_in[0];
  const float* h   = (const float*)d_in[1];
  const float* ea  = (const float*)d_in[2];
  const float* We1 = (const float*)d_in[3];
  const float* be1 = (const float*)d_in[4];
  const float* We2 = (const float*)d_in[5];
  const float* be2 = (const float*)d_in[6];
  const float* Wh1 = (const float*)d_in[7];
  const float* bh1 = (const float*)d_in[8];
  const float* Wh2 = (const float*)d_in[9];
  const float* bh2 = (const float*)d_in[10];
  const float* Wa  = (const float*)d_in[11];
  const float* ba  = (const float*)d_in[12];
  const long long* eidx = (const long long*)d_in[13];

  __bf16* ws   = (__bf16*)d_ws;               // ~205 KB of bf16 weights
  __bf16* W1b  = ws;
  __bf16* W2b  = W1b + 128 * KE;
  __bf16* Wh1b = W2b + 128 * 128;
  __bf16* Wh2b = Wh1b + 128 * KN;
  float* out = (float*)d_out;                 // doubles as message_sum accumulator

  const int wtot = 128 * KE + 2 * 128 * 128 + 128 * KN;
  prep_weights<<<(wtot + 255) / 256, 256, 0, stream>>>(We1, We2, Wh1, Wh2, ws);
  zero_out<<<1024, 256, 0, stream>>>(out, N_NODES * D_H / 4);
  egnn_edge_kernel<<<N_EDGES / EB, EB, 0, stream>>>(x, h, ea, be1, be2, Wa, ba, eidx,
                                                    W1b, W2b, out);
  egnn_node_kernel<<<512, 128, 0, stream>>>(h, bh1, bh2, Wh1b, Wh2b, out);
}
